// Block_11295763989069
// MI455X (gfx1250) — matrix operand order repaired and checked by reading, not compiled, not run on hardware
//
#include <hip/hip_runtime.h>

#define D_EMBD 1024
#define NHEAD  16
#define DHEAD  64
#define SEQ    2048
#define BATCH  4
#define ROWS   (BATCH * SEQ)   // 8192
#define FFDIM  4096

typedef __attribute__((ext_vector_type(16))) __bf16 v16bf;
typedef __attribute__((ext_vector_type(8)))  float  v8f;

union Frag16 {
  v16bf v;
  unsigned int u[8];
  unsigned short h[16];
};

__device__ __forceinline__ unsigned short f2bf(float f) {
  unsigned int u = __builtin_bit_cast(unsigned int, f);
  u += 0x7FFFu + ((u >> 16) & 1u);           // round-to-nearest-even
  return (unsigned short)(u >> 16);
}

__device__ __forceinline__ float gelu_new_f(float x) {
  const float c = 0.7978845608028654f;        // sqrt(2/pi)
  float x3 = x * x * x;
  return 0.5f * x * (1.0f + tanhf(c * (x + 0.044715f * x3)));
}

// ---------------------------------------------------------------- convert
__global__ __launch_bounds__(256)
void f32_to_bf16_kernel(const float* __restrict__ in,
                        unsigned short* __restrict__ out, int n) {
  int i = blockIdx.x * 256 + threadIdx.x;
  if (i < n) out[i] = f2bf(in[i]);
}

// ---------------------------------------------------------------- layernorm
// one block (256 threads) per row of D_EMBD=1024 fp32; bf16 output
__global__ __launch_bounds__(256)
void ln_bf16_kernel(const float* __restrict__ x, const float* __restrict__ g,
                    const float* __restrict__ b, unsigned short* __restrict__ out) {
  __shared__ float red[16];
  const int row = blockIdx.x;
  const int tid = threadIdx.x;
  const float* xr = x + (size_t)row * D_EMBD;
  float4 xv = *(const float4*)(xr + tid * 4);
  float s  = xv.x + xv.y + xv.z + xv.w;
  float sq = xv.x * xv.x + xv.y * xv.y + xv.z * xv.z + xv.w * xv.w;
  #pragma unroll
  for (int m = 1; m < 32; m <<= 1) {
    s  += __shfl_xor(s,  m, 32);
    sq += __shfl_xor(sq, m, 32);
  }
  const int wid = tid >> 5, lane = tid & 31;
  if (lane == 0) { red[wid] = s; red[8 + wid] = sq; }
  __syncthreads();
  if (tid == 0) {
    float ts = 0.f, tq = 0.f;
    for (int i = 0; i < 8; i++) { ts += red[i]; tq += red[8 + i]; }
    red[0] = ts; red[1] = tq;
  }
  __syncthreads();
  const float mu   = red[0] * (1.0f / D_EMBD);
  const float var  = red[1] * (1.0f / D_EMBD) - mu * mu;
  const float rstd = rsqrtf(var + 1e-5f);
  const int i0 = tid * 4;
  float vals[4] = {xv.x, xv.y, xv.z, xv.w};
  #pragma unroll
  for (int e = 0; e < 4; e++) {
    float y = (vals[e] - mu) * rstd * g[i0 + e] + b[i0 + e];
    out[(size_t)row * D_EMBD + i0 + e] = f2bf(y);
  }
}

// ---------------------------------------------------------------- GEMM
// C[M,N] = A[M,K](bf16) @ B[K,N](bf16) + bias, epilogue per MODE:
//   0: Cb = bf16(acc+bias)
//   1: Cf = acc+bias+resid          (fp32)
//   2: Cb = bf16(gelu_new(acc+bias))
// BM=BN=128, BK=32; 8 waves, wave tile 32x64 (2x4 WMMA accumulators).
template<int MODE>
__global__ __launch_bounds__(256)
void gemm_bf16_kernel(const unsigned short* __restrict__ A,
                      const unsigned short* __restrict__ B,
                      const float* __restrict__ bias,
                      const float* __restrict__ resid,
                      float* __restrict__ Cf,
                      unsigned short* __restrict__ Cb,
                      int M, int N, int K) {
  __shared__ unsigned short As[128 * 32];   // [m][k]
  __shared__ unsigned short Bs[128 * 32];   // transposed: [n][k]

  const int tid  = threadIdx.x;
  const int lane = tid & 31, wid = tid >> 5;
  const int half = lane >> 4, lm = lane & 15;
  const int wm = wid & 3, wn = wid >> 2;
  const int m0 = blockIdx.y * 128;
  const int n0 = blockIdx.x * 128;

  const v8f vzero = {};
  v8f acc[2][4];
  #pragma unroll
  for (int i = 0; i < 2; i++)
    #pragma unroll
    for (int j = 0; j < 4; j++) acc[i][j] = vzero;

  const int la_m = tid >> 1;
  const int la_k = (tid & 1) * 16;
  const int lb_k = tid >> 3;
  const int lb_n = (tid & 7) * 16;

  for (int kb = 0; kb < K; kb += 32) {
    uint4 a0, a1, b0, b1;
    {
      const uint4* ag = (const uint4*)(A + (size_t)(m0 + la_m) * K + kb + la_k);
      a0 = ag[0]; a1 = ag[1];
      const uint4* bg = (const uint4*)(B + (size_t)(kb + lb_k) * N + n0 + lb_n);
      b0 = bg[0]; b1 = bg[1];
    }
    __syncthreads();                       // previous tile fully consumed
    *(uint4*)&As[la_m * 32 + la_k]     = a0;
    *(uint4*)&As[la_m * 32 + la_k + 8] = a1;
    {
      unsigned short tmp[16];
      *(uint4*)&tmp[0] = b0; *(uint4*)&tmp[8] = b1;
      #pragma unroll
      for (int e = 0; e < 16; e++) Bs[(lb_n + e) * 32 + lb_k] = tmp[e];
    }
    __syncthreads();

    Frag16 af[2];
    #pragma unroll
    for (int fm = 0; fm < 2; fm++) {
      const int m = wm * 32 + fm * 16 + lm;
      #pragma unroll
      for (int p = 0; p < 8; p++) {
        const int kk = ((p >= 4) ? 16 : 0) + half * 8 + ((2 * p) & 7);
        af[fm].u[p] = *(const unsigned int*)&As[m * 32 + kk];
      }
    }
    #pragma unroll
    for (int fn = 0; fn < 4; fn++) {
      Frag16 bfr;
      const int n = wn * 64 + fn * 16 + lm;
      #pragma unroll
      for (int p = 0; p < 8; p++) {
        const int kk = ((p >= 4) ? 16 : 0) + half * 8 + ((2 * p) & 7);
        bfr.u[p] = *(const unsigned int*)&Bs[n * 32 + kk];
      }
      #pragma unroll
      for (int fm = 0; fm < 2; fm++) {
        acc[fm][fn] = __builtin_amdgcn_wmma_f32_16x16x32_bf16(
            false, af[fm].v, false, bfr.v, (short)0, acc[fm][fn], false, false);
      }
    }
  }

  #pragma unroll
  for (int fm = 0; fm < 2; fm++)
    #pragma unroll
    for (int fn = 0; fn < 4; fn++)
      #pragma unroll
      for (int r = 0; r < 8; r++) {
        const int row = m0 + wm * 32 + fm * 16 + r + 8 * half;
        const int col = n0 + wn * 64 + fn * 16 + lm;
        const size_t idx = (size_t)row * N + col;
        float v = acc[fm][fn][r] + bias[col];
        if (MODE == 0)      Cb[idx] = f2bf(v);
        else if (MODE == 1) Cf[idx] = v + resid[idx];
        else                Cb[idx] = f2bf(gelu_new_f(v));
      }
}

// ---------------------------------------------------------------- attention
// flash-attention, one block per (b,h,128-row q tile); wave owns 16 q rows.
__global__ __launch_bounds__(256)
void attention_kernel(const unsigned short* __restrict__ qkv,
                      unsigned short* __restrict__ out) {
  __shared__ unsigned short Klds[64 * 64];      // [key][d]
  __shared__ unsigned short Vt[64 * 64];        // [d][key]
  __shared__ unsigned short Plds[8][16 * 64];   // per-wave P staging

  const int tid  = threadIdx.x;
  const int lane = tid & 31, wid = tid >> 5;
  const int half = lane >> 4, lm = lane & 15;

  const int qt = blockIdx.x & 15;
  const int bh = blockIdx.x >> 4;
  const int h  = bh & 15;
  const int b  = bh >> 4;
  const int qbase = qt * 128;
  const size_t batch_off = (size_t)b * SEQ * 3072;

  // persistent Q fragments (A-layout gather, d split into two K=32 chunks)
  Frag16 aq[2];
  {
    const int qrow = qbase + wid * 16 + lm;
    const unsigned short* qp = qkv + batch_off + (size_t)qrow * 3072 + h * 64;
    #pragma unroll
    for (int j = 0; j < 2; j++)
      #pragma unroll
      for (int p = 0; p < 8; p++) {
        const int d0 = j * 32 + ((p >= 4) ? 16 : 0) + half * 8 + ((2 * p) & 7);
        aq[j].u[p] = *(const unsigned int*)(qp + d0);
      }
  }

  const v8f vzero = {};
  v8f O[4];
  #pragma unroll
  for (int t = 0; t < 4; t++) O[t] = vzero;
  float mrow[8], lrow[8];
  #pragma unroll
  for (int r = 0; r < 8; r++) { mrow[r] = -3.0e38f; lrow[r] = 0.0f; }

  const int nkb  = qbase / 64 + 2;      // causal: keys <= qbase+127
  const int lkey = tid >> 2;
  const int ld0  = (tid & 3) * 16;

  for (int kb = 0; kb < nkb; kb++) {
    const unsigned short* kp =
        qkv + batch_off + (size_t)(kb * 64 + lkey) * 3072 + 1024 + h * 64 + ld0;
    const unsigned short* vp = kp + 1024;
    uint4 k0 = *(const uint4*)kp, k1 = *(const uint4*)(kp + 8);
    uint4 v0 = *(const uint4*)vp, v1 = *(const uint4*)(vp + 8);
    __syncthreads();                    // prior iteration's LDS reads done
    *(uint4*)&Klds[lkey * 64 + ld0]     = k0;
    *(uint4*)&Klds[lkey * 64 + ld0 + 8] = k1;
    {
      unsigned short tmp[16];
      *(uint4*)&tmp[0] = v0; *(uint4*)&tmp[8] = v1;
      #pragma unroll
      for (int e = 0; e < 16; e++) Vt[(ld0 + e) * 64 + lkey] = tmp[e];
    }
    __syncthreads();

    // S = Q @ K^T  (4 n-tiles of 16 keys, K-dim 64 = 2 chained WMMAs)
    v8f sc[4];
    #pragma unroll
    for (int t = 0; t < 4; t++) {
      v8f c = vzero;
      #pragma unroll
      for (int j = 0; j < 2; j++) {
        Frag16 bk;
        #pragma unroll
        for (int p = 0; p < 8; p++) {
          const int dd = j * 32 + ((p >= 4) ? 16 : 0) + half * 8 + ((2 * p) & 7);
          bk.u[p] = *(const unsigned int*)&Klds[(t * 16 + lm) * 64 + dd];
        }
        c = __builtin_amdgcn_wmma_f32_16x16x32_bf16(
            false, aq[j].v, false, bk.v, (short)0, c, false, false);
      }
      sc[t] = c;
    }

    // scale + causal mask (matches reference: -10000 bias then softmax)
    const int rowb = qbase + wid * 16 + 8 * half;
    #pragma unroll
    for (int t = 0; t < 4; t++) {
      const int col = kb * 64 + t * 16 + lm;
      #pragma unroll
      for (int r = 0; r < 8; r++) {
        const float sv = sc[t][r] * 0.125f;
        sc[t][r] = (col <= rowb + r) ? sv : -10000.0f;
      }
    }

    // online softmax: row max, exp, row sum (xor-reduce within 16-lane group)
    float nm[8], al[8];
    #pragma unroll
    for (int r = 0; r < 8; r++) {
      float tm = fmaxf(fmaxf(sc[0][r], sc[1][r]), fmaxf(sc[2][r], sc[3][r]));
      #pragma unroll
      for (int m = 1; m < 16; m <<= 1) tm = fmaxf(tm, __shfl_xor(tm, m, 32));
      nm[r] = fmaxf(mrow[r], tm);
      al[r] = __expf(mrow[r] - nm[r]);
      mrow[r] = nm[r];
    }
    #pragma unroll
    for (int r = 0; r < 8; r++) {
      float rs = 0.0f;
      #pragma unroll
      for (int t = 0; t < 4; t++) {
        const float p = __expf(sc[t][r] - nm[r]);
        sc[t][r] = p; rs += p;
      }
      #pragma unroll
      for (int m = 1; m < 16; m <<= 1) rs += __shfl_xor(rs, m, 32);
      lrow[r] = lrow[r] * al[r] + rs;
    }
    #pragma unroll
    for (int t = 0; t < 4; t++)
      #pragma unroll
      for (int r = 0; r < 8; r++) O[t][r] *= al[r];

    // stage P (C-layout -> LDS) then gather as A-layout for P @ V
    #pragma unroll
    for (int t = 0; t < 4; t++)
      #pragma unroll
      for (int r = 0; r < 8; r++)
        Plds[wid][(r + 8 * half) * 64 + t * 16 + lm] = f2bf(sc[t][r]);
    __syncthreads();

    Frag16 ap[2];
    #pragma unroll
    for (int j = 0; j < 2; j++)
      #pragma unroll
      for (int p = 0; p < 8; p++) {
        const int k0i = j * 32 + ((p >= 4) ? 16 : 0) + half * 8 + ((2 * p) & 7);
        ap[j].u[p] = *(const unsigned int*)&Plds[wid][lm * 64 + k0i];
      }
    #pragma unroll
    for (int dt = 0; dt < 4; dt++) {
      #pragma unroll
      for (int j = 0; j < 2; j++) {
        Frag16 bv;
        #pragma unroll
        for (int p = 0; p < 8; p++) {
          const int kk = j * 32 + ((p >= 4) ? 16 : 0) + half * 8 + ((2 * p) & 7);
          bv.u[p] = *(const unsigned int*)&Vt[(dt * 16 + lm) * 64 + kk];
        }
        O[dt] = __builtin_amdgcn_wmma_f32_16x16x32_bf16(
            false, ap[j].v, false, bv.v, (short)0, O[dt], false, false);
      }
    }
  }

  const int rowb = qbase + wid * 16 + 8 * half;
  #pragma unroll
  for (int dt = 0; dt < 4; dt++)
    #pragma unroll
    for (int r = 0; r < 8; r++) {
      const float o = O[dt][r] / lrow[r];
      out[((size_t)b * SEQ + rowb + r) * D_EMBD + h * 64 + dt * 16 + lm] = f2bf(o);
    }
}

// ---------------------------------------------------------------- launch
extern "C" void kernel_launch(void* const* d_in, const int* in_sizes, int n_in,
                              void* d_out, int out_size, void* d_ws, size_t ws_size,
                              hipStream_t stream) {
  const float* hidden      = (const float*)d_in[0];
  const float* ln1_g       = (const float*)d_in[1];
  const float* ln1_b       = (const float*)d_in[2];
  const float* w_attn      = (const float*)d_in[3];
  const float* b_attn      = (const float*)d_in[4];
  const float* w_attn_proj = (const float*)d_in[5];
  const float* b_attn_proj = (const float*)d_in[6];
  const float* ln2_g       = (const float*)d_in[7];
  const float* ln2_b       = (const float*)d_in[8];
  const float* w_fc        = (const float*)d_in[9];
  const float* b_fc        = (const float*)d_in[10];
  const float* w_mlp_proj  = (const float*)d_in[11];
  const float* b_mlp_proj  = (const float*)d_in[12];

  char* ws = (char*)d_ws;
  size_t off = 0;
  auto take = [&](size_t bytes) -> char* {
    char* p = ws + off;
    off += (bytes + 255) & ~(size_t)255;
    return p;
  };

  unsigned short* wq  = (unsigned short*)take((size_t)D_EMBD * 3 * D_EMBD * 2);
  unsigned short* wp  = (unsigned short*)take((size_t)D_EMBD * D_EMBD * 2);
  unsigned short* wf  = (unsigned short*)take((size_t)D_EMBD * FFDIM * 2);
  unsigned short* wm  = (unsigned short*)take((size_t)FFDIM * D_EMBD * 2);
  unsigned short* xln = (unsigned short*)take((size_t)ROWS * D_EMBD * 2);   // ln1 out, reused for ln2 out
  unsigned short* big = (unsigned short*)take((size_t)ROWS * FFDIM * 2);    // qkv (3072 cols) then gelu act (4096)
  unsigned short* att = (unsigned short*)take((size_t)ROWS * D_EMBD * 2);
  float*          h1  = (float*)take((size_t)ROWS * D_EMBD * 4);

  auto cvt = [&](const float* src, unsigned short* dst, int n) {
    f32_to_bf16_kernel<<<(n + 255) / 256, 256, 0, stream>>>(src, dst, n);
  };
  cvt(w_attn,      wq, D_EMBD * 3 * D_EMBD);
  cvt(w_attn_proj, wp, D_EMBD * D_EMBD);
  cvt(w_fc,        wf, D_EMBD * FFDIM);
  cvt(w_mlp_proj,  wm, FFDIM * D_EMBD);

  ln_bf16_kernel<<<ROWS, 256, 0, stream>>>(hidden, ln1_g, ln1_b, xln);

  gemm_bf16_kernel<0><<<dim3(3 * D_EMBD / 128, ROWS / 128), 256, 0, stream>>>(
      xln, wq, b_attn, nullptr, nullptr, big, ROWS, 3 * D_EMBD, D_EMBD);

  attention_kernel<<<BATCH * NHEAD * (SEQ / 128), 256, 0, stream>>>(big, att);

  gemm_bf16_kernel<1><<<dim3(D_EMBD / 128, ROWS / 128), 256, 0, stream>>>(
      att, wp, b_attn_proj, hidden, h1, nullptr, ROWS, D_EMBD, D_EMBD);

  ln_bf16_kernel<<<ROWS, 256, 0, stream>>>(h1, ln2_g, ln2_b, xln);

  gemm_bf16_kernel<2><<<dim3(FFDIM / 128, ROWS / 128), 256, 0, stream>>>(
      xln, wf, b_fc, nullptr, nullptr, big, ROWS, FFDIM, D_EMBD);

  gemm_bf16_kernel<1><<<dim3(D_EMBD / 128, ROWS / 128), 256, 0, stream>>>(
      big, wm, b_mlp_proj, h1, (float*)d_out, nullptr, ROWS, D_EMBD, FFDIM);
}
